// Phase2Block_25108378812939
// MI455X (gfx1250) — compile-verified, matrix operand order checked by reading, not run on hardware
//
#include <hip/hip_runtime.h>
#include <hip/hip_bf16.h>
#include <stdint.h>

// ---------------- problem sizes ----------------
#define NB   4
#define NS   2048
#define ND   1024
#define NH   8
#define NHD  64
#define NFF  4096
#define NM   (NB * NS)          // 8192 rows
#define ETA_C   0.1f
#define DT_C    0.01f
#define EPS_C   1e-5f

typedef __bf16 v16bf __attribute__((ext_vector_type(16)));
typedef float  v8f   __attribute__((ext_vector_type(8)));

union FragU { unsigned u[8]; v16bf v; };

#define LDS_STRIDE 20   // dwords per 16-dword tile row: 16B-aligned + conflict-spread

__device__ __forceinline__ unsigned short f2bf(float f) {
  unsigned u = __float_as_uint(f);
  u += 0x7FFFu + ((u >> 16) & 1u);      // round-to-nearest-even
  return (unsigned short)(u >> 16);
}

__device__ __forceinline__ v8f vzero() {
  v8f z;
#pragma unroll
  for (int e = 0; e < 8; ++e) z[e] = 0.0f;
  return z;
}

// Gather a 16x32 bf16 A/B fragment from row-major LDS (K contiguous in row).
// row = 16-row-block base + (lane&15); koff32 = (lane>>4)*4 + kblock*16 (u32).
__device__ __forceinline__ void gatherRM(FragU& f, const unsigned short* lds,
                                         int row, int rs16, int koff32) {
  const unsigned* p = (const unsigned*)lds + row * (rs16 >> 1) + koff32;
  f.u[0] = p[0]; f.u[1] = p[1]; f.u[2] = p[2];  f.u[3] = p[3];
  f.u[4] = p[8]; f.u[5] = p[9]; f.u[6] = p[10]; f.u[7] = p[11];
}

// Gather A fragment of K~^T (A(m=d,k=j) = ldsK[j*64+d]) -- strided u16 reads.
// d = 16-block base + (lane&15); kbe = (lane>>4)*8 (element-unit K offset).
__device__ __forceinline__ void gatherKT(FragU& f, const unsigned short* ldsK,
                                         int d, int kbe) {
#pragma unroll
  for (int v = 0; v < 8; ++v) {
    int e0 = 2 * v, e1 = 2 * v + 1;
    int k0 = ((e0 < 8) ? e0 : e0 + 8) + kbe;
    int k1 = ((e1 < 8) ? e1 : e1 + 8) + kbe;
    unsigned lo = ldsK[k0 * 64 + d];
    unsigned hi = ldsK[k1 * 64 + d];
    f.u[v] = lo | (hi << 16);
  }
}

// Issue async DMA of one 128x32(bf16) tile pair slice into LDS (ASYNCcnt).
__device__ __forceinline__ void prefetch_tiles(
    const unsigned* __restrict__ A32, const unsigned* __restrict__ B32,
    unsigned* la, unsigned* lb, int gm0, int gn0, int K2, int khalf, int t)
{
#pragma unroll
  for (int p = 0; p < 2; ++p) {
    int idx = t + p * 256;               // 0..511
    int row = idx >> 2;
    int c4  = (idx & 3) * 4;
    unsigned la_off = (unsigned)(uintptr_t)(&la[row * LDS_STRIDE + c4]);
    unsigned long long ga =
        (unsigned long long)(uintptr_t)(A32 + (size_t)(gm0 + row) * K2 + khalf + c4);
    asm volatile("global_load_async_to_lds_b128 %0, %1, off"
                 :: "v"(la_off), "v"(ga) : "memory");
    unsigned lb_off = (unsigned)(uintptr_t)(&lb[row * LDS_STRIDE + c4]);
    unsigned long long gb =
        (unsigned long long)(uintptr_t)(B32 + (size_t)(gn0 + row) * K2 + khalf + c4);
    asm volatile("global_load_async_to_lds_b128 %0, %1, off"
                 :: "v"(lb_off), "v"(gb) : "memory");
  }
}

// ============================================================
// bf16 WMMA GEMM:  out(M x Nc) = A(M x K, bf16) * Bt(Nc x K, bf16)^T + bias
// mode 0: outF = acc + bias ; mode 1: outH = bf16(gelu(acc+bias)) ;
// mode 2: outF = res + acc + bias
// 256 threads (8 waves), 128x128 tile; wave = 32x64 (2x4 WMMA tiles).
// Double-buffered LDS filled by GLOBAL_LOAD_ASYNC_TO_LDS_B128.
// ============================================================
__global__ __launch_bounds__(256) void gemm_bf16_kernel(
    const unsigned short* __restrict__ A, const unsigned short* __restrict__ Bt,
    const float* __restrict__ bias, const float* __restrict__ res,
    float* __restrict__ outF, unsigned short* __restrict__ outH,
    int M, int Ncols, int K, int mode)
{
  __shared__ unsigned lA[2][128 * LDS_STRIDE];
  __shared__ unsigned lB[2][128 * LDS_STRIDE];

  const int t    = threadIdx.x;
  const int lane = t & 31;
  const int wv   = t >> 5;
  const int mw   = (wv >> 1) * 32;
  const int nw   = (wv & 1) * 64;
  const int gm0  = blockIdx.y * 128;
  const int gn0  = blockIdx.x * 128;
  const int K2   = K >> 1;
  const unsigned* A32 = (const unsigned*)A;
  const unsigned* B32 = (const unsigned*)Bt;

  v8f acc[2][4];
#pragma unroll
  for (int i = 0; i < 2; ++i)
#pragma unroll
    for (int j = 0; j < 4; ++j) acc[i][j] = vzero();

  const int lr  = lane & 15;
  const int kb2 = (lane >> 4) * 4;

  const int nk = K >> 5;
  prefetch_tiles(A32, B32, &lA[0][0], &lB[0][0], gm0, gn0, K2, 0, t);

  for (int kk = 0; kk < nk; ++kk) {
    const int cur = kk & 1;
    if (kk + 1 < nk) {
      prefetch_tiles(A32, B32, &lA[cur ^ 1][0], &lB[cur ^ 1][0],
                     gm0, gn0, K2, (kk + 1) * 16, t);
      asm volatile("s_wait_asynccnt 0x4" ::: "memory");
    } else {
      asm volatile("s_wait_asynccnt 0x0" ::: "memory");
    }
    __syncthreads();

    FragU fa[2], fb[4];
#pragma unroll
    for (int mi = 0; mi < 2; ++mi) {
      const unsigned* p = &lA[cur][(mw + mi * 16 + lr) * LDS_STRIDE + kb2];
      fa[mi].u[0] = p[0]; fa[mi].u[1] = p[1]; fa[mi].u[2] = p[2];  fa[mi].u[3] = p[3];
      fa[mi].u[4] = p[8]; fa[mi].u[5] = p[9]; fa[mi].u[6] = p[10]; fa[mi].u[7] = p[11];
    }
#pragma unroll
    for (int ni = 0; ni < 4; ++ni) {
      const unsigned* p = &lB[cur][(nw + ni * 16 + lr) * LDS_STRIDE + kb2];
      fb[ni].u[0] = p[0]; fb[ni].u[1] = p[1]; fb[ni].u[2] = p[2];  fb[ni].u[3] = p[3];
      fb[ni].u[4] = p[8]; fb[ni].u[5] = p[9]; fb[ni].u[6] = p[10]; fb[ni].u[7] = p[11];
    }

#pragma unroll
    for (int mi = 0; mi < 2; ++mi)
#pragma unroll
      for (int ni = 0; ni < 4; ++ni)
        acc[mi][ni] = __builtin_amdgcn_wmma_f32_16x16x32_bf16(
            false, fa[mi].v, false, fb[ni].v, (short)0, acc[mi][ni], false, false);
    __syncthreads();
  }

  const int mo = (lane >> 4) * 8;
#pragma unroll
  for (int mi = 0; mi < 2; ++mi) {
#pragma unroll
    for (int ni = 0; ni < 4; ++ni) {
      int n     = gn0 + nw + ni * 16 + lr;
      int mbase = gm0 + mw + mi * 16 + mo;
      float bv  = bias[n];
#pragma unroll
      for (int r = 0; r < 8; ++r) {
        size_t o  = (size_t)(mbase + r) * Ncols + n;
        float val = acc[mi][ni][r] + bv;
        if (mode == 2) val += res[o];
        if (mode == 1) {
          float c = val * val * val;
          float u = 0.7978845608028654f * (val + 0.044715f * c);
          val = 0.5f * val * (1.0f + tanhf(u));
          outH[o] = f2bf(val);
        } else {
          outF[o] = val;
        }
      }
    }
  }
}

// ============================================================
// Weight convert+transpose: w(K x Nc, f32) -> wT(Nc x K, bf16)
// ============================================================
__global__ __launch_bounds__(256) void convT_kernel(
    const float* __restrict__ w, unsigned short* __restrict__ wT, int K, int Ncols)
{
  size_t idx = (size_t)blockIdx.x * 256 + threadIdx.x;
  if (idx >= (size_t)K * Ncols) return;
  int k = (int)(idx / Ncols);
  int n = (int)(idx % Ncols);
  wT[(size_t)n * K + k] = f2bf(w[idx]);
}

// ============================================================
// LN1 + potential head: one block per row.
// ============================================================
__global__ __launch_bounds__(256) void ln1_pot_kernel(
    const float* __restrict__ x, const float* __restrict__ g1, const float* __restrict__ b1,
    const float* __restrict__ w_pot, const float* __restrict__ b_pot,
    float* __restrict__ gammaArr, float2* __restrict__ dC)
{
  __shared__ float r0[256], r1[256];
  int row = blockIdx.x;
  const float* xr = x + (size_t)row * ND;
  float s = 0.f, s2 = 0.f;
  for (int d = threadIdx.x; d < ND; d += 256) { float v = xr[d]; s += v; s2 += v * v; }
  r0[threadIdx.x] = s; r1[threadIdx.x] = s2; __syncthreads();
  for (int o = 128; o > 0; o >>= 1) {
    if (threadIdx.x < o) { r0[threadIdx.x] += r0[threadIdx.x + o]; r1[threadIdx.x] += r1[threadIdx.x + o]; }
    __syncthreads();
  }
  float mean = r0[0] * (1.0f / ND);
  float var  = r1[0] * (1.0f / ND) - mean * mean;
  float rstd = rsqrtf(var + EPS_C);
  __syncthreads();

  float p0 = 0.f, p1 = 0.f;
  for (int d = threadIdx.x; d < ND; d += 256) {
    float xn = (xr[d] - mean) * rstd * g1[d] + b1[d];
    p0 += xn * w_pot[2 * d];
    p1 += xn * w_pot[2 * d + 1];
  }
  r0[threadIdx.x] = p0; r1[threadIdx.x] = p1; __syncthreads();
  for (int o = 128; o > 0; o >>= 1) {
    if (threadIdx.x < o) { r0[threadIdx.x] += r0[threadIdx.x + o]; r1[threadIdx.x] += r1[threadIdx.x + o]; }
    __syncthreads();
  }
  if (threadIdx.x == 0) {
    float V  = r0[0] + b_pot[0];
    float tt = r1[0] + b_pot[1];
    float sp = (tt > 20.f) ? tt : log1pf(__expf(tt));
    float gm = sp + 0.1f;
    gammaArr[row] = gm;
    dC[row] = make_float2(V + 2.0f, -gm);
  }
}

// ============================================================
// Continued-fraction scans: 8 independent length-2048 complex recurrences.
// ============================================================
__global__ void cf_scan_kernel(const float2* __restrict__ dC,
                               float2* __restrict__ aF, float2* __restrict__ bB)
{
  int t = threadIdx.x;
  if (t >= 2 * NB) return;
  int b = t >> 1, dir = t & 1;
  if (dir == 0) {
    float2 c = dC[b * NS];
    aF[b * NS] = c;
    for (int n = 1; n < NS; ++n) {
      float2 dn = dC[b * NS + n];
      float den = c.x * c.x + c.y * c.y;
      c.x = dn.x - c.x / den;
      c.y = dn.y + c.y / den;
      aF[b * NS + n] = c;
    }
  } else {
    float2 c = dC[b * NS + NS - 1];
    bB[b * NS + NS - 1] = c;
    for (int n = NS - 2; n >= 0; --n) {
      float2 dn = dC[b * NS + n];
      float den = c.x * c.x + c.y * c.y;
      c.x = dn.x - c.x / den;
      c.y = dn.y + c.y / den;
      bB[b * NS + n] = c;
    }
  }
}

// ============================================================
// BK residual add + LN2: x1 = x + f @ w_bk + b_bk ; xn2 = bf16(LN2(x1))
// ============================================================
__global__ __launch_bounds__(256) void bk_ln2_kernel(
    const float* __restrict__ x,
    const float2* __restrict__ aF, const float2* __restrict__ bB, const float2* __restrict__ dC,
    const float* __restrict__ w_bk, const float* __restrict__ b_bk,
    const float* __restrict__ g2, const float* __restrict__ be2,
    float* __restrict__ x1, unsigned short* __restrict__ xn2)
{
  __shared__ float r0[256], r1[256];
  int row = blockIdx.x;
  float2 a = aF[row], bb = bB[row], dd = dC[row];
  float sx = a.x + bb.x - dd.x, sy = a.y + bb.y - dd.y;
  float den = sx * sx + sy * sy;
  float f0 = sx / den, f1 = -sy / den;

  const float* xr = x + (size_t)row * ND;
  float* x1r = x1 + (size_t)row * ND;
  float s = 0.f, s2 = 0.f;
  for (int d = threadIdx.x; d < ND; d += 256) {
    float v = xr[d] + f0 * w_bk[d] + f1 * w_bk[ND + d] + b_bk[d];
    x1r[d] = v; s += v; s2 += v * v;
  }
  r0[threadIdx.x] = s; r1[threadIdx.x] = s2; __syncthreads();
  for (int o = 128; o > 0; o >>= 1) {
    if (threadIdx.x < o) { r0[threadIdx.x] += r0[threadIdx.x + o]; r1[threadIdx.x] += r1[threadIdx.x + o]; }
    __syncthreads();
  }
  float mean = r0[0] * (1.0f / ND);
  float var  = r1[0] * (1.0f / ND) - mean * mean;
  float rstd = rsqrtf(var + EPS_C);
  unsigned short* o2 = xn2 + (size_t)row * ND;
  for (int d = threadIdx.x; d < ND; d += 256)
    o2[d] = f2bf((x1r[d] - mean) * rstd * g2[d] + be2[d]);
}

// ============================================================
// Generic row LN -> bf16 (used for LN3)
// ============================================================
__global__ __launch_bounds__(256) void ln_bf16_kernel(
    const float* __restrict__ xin, const float* __restrict__ g, const float* __restrict__ be,
    unsigned short* __restrict__ out)
{
  __shared__ float r0[256], r1[256];
  int row = blockIdx.x;
  const float* xr = xin + (size_t)row * ND;
  float s = 0.f, s2 = 0.f;
  for (int d = threadIdx.x; d < ND; d += 256) { float v = xr[d]; s += v; s2 += v * v; }
  r0[threadIdx.x] = s; r1[threadIdx.x] = s2; __syncthreads();
  for (int o = 128; o > 0; o >>= 1) {
    if (threadIdx.x < o) { r0[threadIdx.x] += r0[threadIdx.x + o]; r1[threadIdx.x] += r1[threadIdx.x + o]; }
    __syncthreads();
  }
  float mean = r0[0] * (1.0f / ND);
  float var  = r1[0] * (1.0f / ND) - mean * mean;
  float rstd = rsqrtf(var + EPS_C);
  unsigned short* ot = out + (size_t)row * ND;
  for (int d = threadIdx.x; d < ND; d += 256)
    ot[d] = f2bf((xr[d] - mean) * rstd * g[d] + be[d]);
}

// ============================================================
// Chunked WMMA Hebbian fast-weight scan (chunk C=32).
//   P_t = prod decay ; Q~ = P_t q ; K~ = k / P_t ; V' = eta v
//   S = tril(Q~ K~^T) ; O = Q~ W0 + S V' ; W = P_C (W0 + K~^T V')
// One block per (b,h); 2 waves split the e(=64) dim; W half kept in
// persistent accumulator VGPRs + bf16 transposed LDS copy for B-frags.
// ============================================================
__global__ __launch_bounds__(64) void hebbian_wmma_kernel(
    const float* __restrict__ qkv, const float* __restrict__ gammaArr,
    unsigned short* __restrict__ heb)
{
  __shared__ float sg[32];
  __shared__ __align__(16) unsigned short ldsQ[32 * 64];
  __shared__ __align__(16) unsigned short ldsK[32 * 64];
  __shared__ __align__(16) unsigned short ldsVt[64 * 32];
  __shared__ __align__(16) unsigned short ldsWT[64 * 64];
  __shared__ __align__(16) unsigned short ldsS[32 * 32];

  const int bh   = blockIdx.x;
  const int b    = bh / NH, h = bh % NH;
  const int t    = threadIdx.x;
  const int w    = t >> 5;
  const int lane = t & 31;
  const int lr   = lane & 15;
  const int kb2  = (lane >> 4) * 4;   // u32 K offset
  const int kbe  = (lane >> 4) * 8;   // element K offset
  const int mo   = (lane >> 4) * 8;   // C/D layout m offset
  const int ebase = w * 32;           // this wave's e-half

  {
    unsigned* wt32 = (unsigned*)ldsWT;
    for (int i = t; i < 64 * 64 / 2; i += 64) wt32[i] = 0u;
  }
  v8f Wacc[4][2];
#pragma unroll
  for (int mi = 0; mi < 4; ++mi)
#pragma unroll
    for (int ni = 0; ni < 2; ++ni) Wacc[mi][ni] = vzero();
  __syncthreads();

  const int j  = t >> 1;    // chunk row 0..31
  const int hf = t & 1;     // column half of the 64-wide head dim

  for (int c0 = 0; c0 < NS; c0 += 32) {
    // ---------- P1: load chunk, decay prefix, build Q~,K~,V' ----------
    size_t rowo = (size_t)(b * NS + c0 + j) * (3 * NH * NHD) + h * NHD + hf * 32;
    float qv[32], kv[32], vv[32];
    const float4* q4 = (const float4*)(qkv + rowo);
    const float4* k4 = (const float4*)(qkv + rowo + NH * NHD);
    const float4* v4 = (const float4*)(qkv + rowo + 2 * NH * NHD);
#pragma unroll
    for (int i = 0; i < 8; ++i) {
      float4 a = q4[i]; qv[4*i] = a.x; qv[4*i+1] = a.y; qv[4*i+2] = a.z; qv[4*i+3] = a.w;
      float4 c = k4[i]; kv[4*i] = c.x; kv[4*i+1] = c.y; kv[4*i+2] = c.z; kv[4*i+3] = c.w;
      float4 d = v4[i]; vv[4*i] = d.x; vv[4*i+1] = d.y; vv[4*i+2] = d.z; vv[4*i+3] = d.w;
    }
    if (hf == 0) sg[j] = gammaArr[b * NS + c0 + j];
    __syncthreads();
    float sum = 0.f, sumAll = 0.f;
#pragma unroll
    for (int i = 0; i < 32; ++i) {
      float gi = sg[i];
      sumAll += gi;
      if (i <= j) sum += gi;
    }
    float Pj  = __expf(-DT_C * sum);
    float iPj = __expf( DT_C * sum);
    float Pc  = __expf(-DT_C * sumAll);

    unsigned* q32 = (unsigned*)&ldsQ[j * 64 + hf * 32];
    unsigned* k32 = (unsigned*)&ldsK[j * 64 + hf * 32];
#pragma unroll
    for (int i = 0; i < 16; ++i) {
      q32[i] = (unsigned)f2bf(qv[2*i] * Pj)  | ((unsigned)f2bf(qv[2*i+1] * Pj)  << 16);
      k32[i] = (unsigned)f2bf(kv[2*i] * iPj) | ((unsigned)f2bf(kv[2*i+1] * iPj) << 16);
    }
#pragma unroll
    for (int i = 0; i < 32; ++i)
      ldsVt[(hf * 32 + i) * 32 + j] = f2bf(vv[i] * ETA_C);
    __syncthreads();

    // ---------- P2: S = tril(Q~ K~^T)  (wave0 only; wave-uniform branch) ----
    if (w == 0) {
      v8f sacc[2][2];
#pragma unroll
      for (int mi = 0; mi < 2; ++mi)
#pragma unroll
        for (int ni = 0; ni < 2; ++ni) sacc[mi][ni] = vzero();
#pragma unroll
      for (int kd = 0; kd < 2; ++kd) {
        FragU aq[2], bk[2];
#pragma unroll
        for (int mi = 0; mi < 2; ++mi) gatherRM(aq[mi], ldsQ, mi * 16 + lr, 64, kb2 + kd * 16);
#pragma unroll
        for (int ni = 0; ni < 2; ++ni) gatherRM(bk[ni], ldsK, ni * 16 + lr, 64, kb2 + kd * 16);
#pragma unroll
        for (int mi = 0; mi < 2; ++mi)
#pragma unroll
          for (int ni = 0; ni < 2; ++ni)
            sacc[mi][ni] = __builtin_amdgcn_wmma_f32_16x16x32_bf16(
                false, aq[mi].v, false, bk[ni].v, (short)0, sacc[mi][ni], false, false);
      }
#pragma unroll
      for (int mi = 0; mi < 2; ++mi)
#pragma unroll
        for (int ni = 0; ni < 2; ++ni) {
          int tr = mi * 16 + mo;
          int jc = ni * 16 + lr;
#pragma unroll
          for (int r = 0; r < 8; ++r) {
            float vlv = (jc <= tr + r) ? sacc[mi][ni][r] : 0.0f;
            ldsS[(tr + r) * 32 + jc] = f2bf(vlv);
          }
        }
    }

    // ---------- O1 = Q~ @ W0 (own e-half) ----------
    v8f oacc[2][2];
#pragma unroll
    for (int mi = 0; mi < 2; ++mi)
#pragma unroll
      for (int ni = 0; ni < 2; ++ni) oacc[mi][ni] = vzero();
#pragma unroll
    for (int kd = 0; kd < 2; ++kd) {
      FragU aq[2], bw[2];
#pragma unroll
      for (int mi = 0; mi < 2; ++mi) gatherRM(aq[mi], ldsQ, mi * 16 + lr, 64, kb2 + kd * 16);
#pragma unroll
      for (int ni = 0; ni < 2; ++ni) gatherRM(bw[ni], ldsWT, ebase + ni * 16 + lr, 64, kb2 + kd * 16);
#pragma unroll
      for (int mi = 0; mi < 2; ++mi)
#pragma unroll
        for (int ni = 0; ni < 2; ++ni)
          oacc[mi][ni] = __builtin_amdgcn_wmma_f32_16x16x32_bf16(
              false, aq[mi].v, false, bw[ni].v, (short)0, oacc[mi][ni], false, false);
    }
    __syncthreads();   // ldsS ready

    // ---------- O2 += S_m @ V' ; store O ----------
    {
      FragU as[2], bv[2];
#pragma unroll
      for (int mi = 0; mi < 2; ++mi) gatherRM(as[mi], ldsS, mi * 16 + lr, 32, kb2);
#pragma unroll
      for (int ni = 0; ni < 2; ++ni) gatherRM(bv[ni], ldsVt, ebase + ni * 16 + lr, 32, kb2);
#pragma unroll
      for (int mi = 0; mi < 2; ++mi)
#pragma unroll
        for (int ni = 0; ni < 2; ++ni)
          oacc[mi][ni] = __builtin_amdgcn_wmma_f32_16x16x32_bf16(
              false, as[mi].v, false, bv[ni].v, (short)0, oacc[mi][ni], false, false);
    }
#pragma unroll
    for (int mi = 0; mi < 2; ++mi)
#pragma unroll
      for (int ni = 0; ni < 2; ++ni) {
        int e = ebase + ni * 16 + lr;
#pragma unroll
        for (int r = 0; r < 8; ++r) {
          int tt = mi * 16 + mo + r;
          heb[(size_t)(b * NS + c0 + tt) * (NH * NHD) + h * NHD + e] = f2bf(oacc[mi][ni][r]);
        }
      }

    // ---------- state update: W = Pc (W0 + K~^T V')  (own e-half) ----------
    {
      FragU akt[4], bv[2];
#pragma unroll
      for (int mi = 0; mi < 4; ++mi) gatherKT(akt[mi], ldsK, mi * 16 + lr, kbe);
#pragma unroll
      for (int ni = 0; ni < 2; ++ni) gatherRM(bv[ni], ldsVt, ebase + ni * 16 + lr, 32, kb2);
      v8f zc = vzero();
#pragma unroll
      for (int mi = 0; mi < 4; ++mi)
#pragma unroll
        for (int ni = 0; ni < 2; ++ni) {
          v8f dw = __builtin_amdgcn_wmma_f32_16x16x32_bf16(
              false, akt[mi].v, false, bv[ni].v, (short)0, zc, false, false);
          int e = ebase + ni * 16 + lr;
          int dbase = mi * 16 + mo;
#pragma unroll
          for (int r = 0; r < 8; ++r) {
            float nw_ = Pc * (Wacc[mi][ni][r] + dw[r]);
            Wacc[mi][ni][r] = nw_;
            ldsWT[e * 64 + dbase + r] = f2bf(nw_);
          }
        }
    }
    __syncthreads();   // chunk complete; safe to rebuild Q/K/V tiles
  }
}

// ============================================================
extern "C" void kernel_launch(void* const* d_in, const int* in_sizes, int n_in,
                              void* d_out, int out_size, void* d_ws, size_t ws_size,
                              hipStream_t stream)
{
  (void)in_sizes; (void)n_in; (void)out_size; (void)ws_size;
  const float* x     = (const float*)d_in[0];
  const float* ln1_g = (const float*)d_in[1];
  const float* ln1_b = (const float*)d_in[2];
  const float* ln2_g = (const float*)d_in[3];
  const float* ln2_b = (const float*)d_in[4];
  const float* ln3_g = (const float*)d_in[5];
  const float* ln3_b = (const float*)d_in[6];
  const float* w_pot = (const float*)d_in[7];
  const float* b_pot = (const float*)d_in[8];
  const float* w_bk  = (const float*)d_in[9];
  const float* b_bk  = (const float*)d_in[10];
  const float* w_qkv = (const float*)d_in[11];
  const float* b_qkv = (const float*)d_in[12];
  const float* w_out = (const float*)d_in[13];
  const float* b_out = (const float*)d_in[14];
  const float* w_ff1 = (const float*)d_in[15];
  const float* b_ff1 = (const float*)d_in[16];
  const float* w_ff2 = (const float*)d_in[17];
  const float* b_ff2 = (const float*)d_in[18];
  float* outp = (float*)d_out;

  char* arena = (char*)d_ws;
  size_t off = 0;
  auto alloc = [&](size_t bytes) -> void* {
    off = (off + 255) & ~(size_t)255;
    void* p = arena + off;
    off += bytes;
    return p;
  };

  unsigned short* wqkvT = (unsigned short*)alloc((size_t)1536 * 1024 * 2);
  unsigned short* woutT = (unsigned short*)alloc((size_t)1024 * 512 * 2);
  unsigned short* wff1T = (unsigned short*)alloc((size_t)4096 * 1024 * 2);
  unsigned short* wff2T = (unsigned short*)alloc((size_t)1024 * 4096 * 2);
  float*          gamA  = (float*)alloc((size_t)NM * 4);
  float2*         dC    = (float2*)alloc((size_t)NM * 8);
  float2*         aF    = (float2*)alloc((size_t)NM * 8);
  float2*         bB    = (float2*)alloc((size_t)NM * 8);
  float*          x1    = (float*)alloc((size_t)NM * ND * 4);
  unsigned short* xn2   = (unsigned short*)alloc((size_t)NM * ND * 2);
  float*          qkv   = (float*)alloc((size_t)NM * 1536 * 4);
  unsigned short* heb   = (unsigned short*)alloc((size_t)NM * 512 * 2);
  float*          x2    = (float*)alloc((size_t)NM * ND * 4);
  unsigned short* xn3   = (unsigned short*)alloc((size_t)NM * ND * 2);
  unsigned short* hff   = (unsigned short*)alloc((size_t)NM * NFF * 2);

  convT_kernel<<<(1024 * 1536 + 255) / 256, 256, 0, stream>>>(w_qkv, wqkvT, 1024, 1536);
  convT_kernel<<<(512 * 1024 + 255) / 256, 256, 0, stream>>>(w_out, woutT, 512, 1024);
  convT_kernel<<<(1024 * 4096 + 255) / 256, 256, 0, stream>>>(w_ff1, wff1T, 1024, 4096);
  convT_kernel<<<(4096 * 1024 + 255) / 256, 256, 0, stream>>>(w_ff2, wff2T, 4096, 1024);

  ln1_pot_kernel<<<NM, 256, 0, stream>>>(x, ln1_g, ln1_b, w_pot, b_pot, gamA, dC);
  cf_scan_kernel<<<1, 8, 0, stream>>>(dC, aF, bB);
  bk_ln2_kernel<<<NM, 256, 0, stream>>>(x, aF, bB, dC, w_bk, b_bk, ln2_g, ln2_b, x1, xn2);

  gemm_bf16_kernel<<<dim3(1536 / 128, NM / 128), 256, 0, stream>>>(
      xn2, wqkvT, b_qkv, nullptr, qkv, nullptr, NM, 1536, 1024, 0);

  hebbian_wmma_kernel<<<NB * NH, 64, 0, stream>>>(qkv, gamA, heb);

  gemm_bf16_kernel<<<dim3(1024 / 128, NM / 128), 256, 0, stream>>>(
      heb, woutT, b_out, x1, x2, nullptr, NM, 1024, 512, 2);

  ln_bf16_kernel<<<NM, 256, 0, stream>>>(x2, ln3_g, ln3_b, xn3);

  gemm_bf16_kernel<<<dim3(4096 / 128, NM / 128), 256, 0, stream>>>(
      xn3, wff1T, b_ff1, nullptr, nullptr, hff, NM, 4096, 1024, 1);

  gemm_bf16_kernel<<<dim3(1024 / 128, NM / 128), 256, 0, stream>>>(
      hff, wff2T, b_ff2, x2, outp, nullptr, NM, 1024, 4096, 2);
}
